// TextureExtractor_55989193671417
// MI455X (gfx1250) — compile-verified, matrix operand order checked by reading, not run on hardware
//
#include <hip/hip_runtime.h>
#include <hip/hip_bf16.h>
#include <cstdint>
#include <cstddef>

typedef __attribute__((ext_vector_type(16))) _Float16 v16h;
typedef __attribute__((ext_vector_type(8)))  float    v8f;

#define B_    8
#define HW_   (256*256)
#define NPIX  (B_*HW_)          // 524288 pixels
#define NSTAT 524288.0f         // B*H*W elements per channel for BN

// ---------------------------------------------------------------- gray + quantize
__global__ void k_gray(const float* __restrict__ img, float* __restrict__ q) {
    int idx = blockIdx.x * 256 + threadIdx.x;            // NPIX exact
    int b = idx >> 16, rem = idx & 65535;
    const float* base = img + (size_t)b * 3 * HW_ + rem;
    float g = 0.299f * base[0] + 0.587f * base[HW_] + 0.116f * base[2 * HW_];
    q[idx] = fminf(fmaxf(floorf(g * 255.0f), 0.0f), 255.0f);
}

// ---------------------------------------------------------------- bilateral (LUT-based, LDS tile)
__global__ void k_bilateral(const float* __restrict__ q, float* __restrict__ filt) {
    __shared__ float tile[46][46];
    __shared__ float lutc[256];   // color weights: d is integer-valued -> exact LUT
    __shared__ float sw1[31];     // separable spatial 1-D weights
    int tx = threadIdx.x, ty = threadIdx.y;
    int tid = ty * 16 + tx;
    int b = blockIdx.z;
    int gx0 = blockIdx.x * 16, gy0 = blockIdx.y * 16;
    for (int i = tid; i < 46 * 46; i += 256) {
        int r = i / 46, c = i % 46;
        int gy = gy0 - 15 + r, gx = gx0 - 15 + c;
        gy = gy < 0 ? -gy : (gy > 255 ? 510 - gy : gy);   // reflect (no edge repeat)
        gx = gx < 0 ? -gx : (gx > 255 ? 510 - gx : gx);
        tile[r][c] = q[((size_t)b * 256 + gy) * 256 + gx];
    }
    lutc[tid] = expf(-0.5f * (float)(tid * tid) / (75.0f * 75.0f));
    if (tid < 31) { int d = tid - 15; sw1[tid] = expf(-0.5f * (float)(d * d) / 225.0f); }
    __syncthreads();

    float c0 = tile[ty + 15][tx + 15];
    float num = 0.0f, den = 0.0f;
    for (int dx = -15; dx <= 15; ++dx) {
        float wx = sw1[dx + 15];
        const float* row = &tile[ty + 15 + dx][tx + 15];
        for (int dy = -15; dy <= 15; ++dy) {
            if (dx * dx + dy * dy > 225) continue;        // disc
            float sh = row[dy];
            int ad = (int)fabsf(sh - c0);                 // exact: integer-valued floats
            float w = wx * sw1[dy + 15] * lutc[ad];
            num = fmaf(w, sh, num);
            den += w;
        }
    }
    int y = gy0 + ty, x = gx0 + tx;
    filt[((size_t)b * 256 + y) * 256 + x] = rintf(num / den);   // jnp.round = half-to-even
}

// ---------------------------------------------------------------- LBP (/255, zero border)
__global__ void k_lbp(const float* __restrict__ filt, float* __restrict__ lbp) {
    int idx = blockIdx.x * 256 + threadIdx.x;            // NPIX exact
    int x = idx & 255, y = (idx >> 8) & 255, b = idx >> 16;
    if (x == 0 || x == 255 || y == 0 || y == 255) { lbp[idx] = 0.0f; return; }
    const float* im = filt + (size_t)b * HW_;
    float c = im[(y << 8) + x];
    float p = 128.0f;                                    // (0,0) offset: always >=
    p += 1.0f  * (float)(im[((y + 1) << 8) + x]       >= c);
    p += 6.0f  * (float)(im[(y << 8) + (x - 1)]       >= c);  // duplicated offset: 2+4
    p += 8.0f  * (float)(im[((y - 1) << 8) + (x - 1)] >= c);
    p += 48.0f * (float)(im[((y - 1) << 8) + x]       >= c);  // duplicated offset: 16+32
    p += 64.0f * (float)(im[(y << 8) + (x + 1)]       >= c);
    lbp[idx] = p * (1.0f / 255.0f);
}

// ---------------------------------------------------------------- conv1 (1->32, 3x3, +bias) -> NHWC f32
__global__ void k_conv1(const float* __restrict__ lbp, const float* __restrict__ w1,
                        const float* __restrict__ b1, float* __restrict__ x1) {
    __shared__ float sw[288];
    __shared__ float sb[32];
    int tid = threadIdx.x;
    for (int i = tid; i < 288; i += 256) sw[i] = w1[i];
    if (tid < 32) sb[tid] = b1[tid];
    __syncthreads();
    int idx = blockIdx.x * 256 + tid;                    // NPIX exact
    int x = idx & 255, y = (idx >> 8) & 255, b = idx >> 16;
    float v[9];
#pragma unroll
    for (int t = 0; t < 9; ++t) {
        int sy = y + t / 3 - 1, sx = x + t % 3 - 1;
        v[t] = (sy >= 0 && sy < 256 && sx >= 0 && sx < 256)
                 ? lbp[((size_t)b << 16) + (sy << 8) + sx] : 0.0f;
    }
    size_t obase = (size_t)idx * 32;
    for (int oc = 0; oc < 32; ++oc) {
        float acc = sb[oc];
#pragma unroll
        for (int t = 0; t < 9; ++t) acc = fmaf(sw[oc * 9 + t], v[t], acc);
        x1[obase + oc] = acc;
    }
}

// ---------------------------------------------------------------- zero BN stat accumulators
__global__ void k_zero_stats(float* __restrict__ s1, float* __restrict__ s2) {
    int t = threadIdx.x;
    if (t < 64)  s1[t] = 0.0f;   // sum[32] + sumsq[32]
    if (t < 128) s2[t] = 0.0f;   // sum[64] + sumsq[64]
}

// ---------------------------------------------------------------- per-channel sum/sumsq (NHWC input)
__global__ void k_stats(const float* __restrict__ x, float* __restrict__ sum,
                        float* __restrict__ sumsq, int C, size_t total) {
    __shared__ float ss[64], sq[64];
    int tid = threadIdx.x;
    if (tid < C) { ss[tid] = 0.0f; sq[tid] = 0.0f; }
    __syncthreads();
    int c = tid & (C - 1);                               // stride multiple of C -> channel fixed
    float s = 0.0f, s2 = 0.0f;
    size_t stride = (size_t)gridDim.x * 256;
    for (size_t i = (size_t)blockIdx.x * 256 + tid; i < total; i += stride) {
        float v = x[i];
        s += v; s2 = fmaf(v, v, s2);
    }
    atomicAdd(&ss[c], s); atomicAdd(&sq[c], s2);
    __syncthreads();
    if (tid < C) { atomicAdd(&sum[tid], ss[tid]); atomicAdd(&sumsq[tid], sq[tid]); }
}

// ---------------------------------------------------------------- BN1 + ReLU -> f16 NHWC
__global__ void k_bnrelu1(const float* __restrict__ x1, const float* __restrict__ g,
                          const float* __restrict__ beta, const float* __restrict__ st,
                          _Float16* __restrict__ x1n) {
    size_t idx = (size_t)blockIdx.x * 256 + threadIdx.x; // NPIX*32 exact
    int c = (int)(idx & 31);
    float mean = st[c] * (1.0f / NSTAT);
    float var  = st[32 + c] * (1.0f / NSTAT) - mean * mean;
    float sc = g[c] * rsqrtf(var + 1e-5f);
    float sh = beta[c] - mean * sc;
    x1n[idx] = (_Float16)fmaxf(fmaf(x1[idx], sc, sh), 0.0f);
}

// ---------------------------------------------------------------- pack w2 into wave-resident WMMA B layout
// layout: w2p[((tap*4 + ntile)*32 + lane)*16 + j]; lane: n = nt*16+(lane&15), hi=lane>>4
// halfs j0..7 -> K = hi*8 + j ; j8..15 -> K = 16 + hi*8 + (j-8)   (mirrors 16-bit A/B layout)
__global__ void k_packw2(const float* __restrict__ w2, _Float16* __restrict__ w2p) {
    int id = blockIdx.x * 256 + threadIdx.x;
    if (id >= 9 * 4 * 32) return;
    int lane = id & 31, nt = (id >> 5) & 3, t = id >> 7;
    int hi = lane >> 4;
    int n = nt * 16 + (lane & 15);
    int ky = t / 3, kx = t % 3;
#pragma unroll
    for (int j = 0; j < 16; ++j) {
        int k = (j < 8) ? (hi * 8 + j) : (16 + hi * 8 + (j - 8));
        w2p[(size_t)id * 16 + j] = (_Float16)w2[(((size_t)n * 32 + k) * 3 + ky) * 3 + kx];
    }
}

// ---------------------------------------------------------------- conv2 via WMMA: 16 px/wave, 64 oc, 9 taps
__global__ void __launch_bounds__(256)
k_conv2_wmma(const _Float16* __restrict__ x1n, const _Float16* __restrict__ w2p,
             const float* __restrict__ b2, float* __restrict__ x2) {
    __shared__ float sb2[64];
    int tid = threadIdx.x;
    if (tid < 64) sb2[tid] = b2[tid];
    __syncthreads();
    int wave = tid >> 5, lane = tid & 31;
    int tile = blockIdx.x * 8 + wave;                    // 32768 tiles exact
    int x0 = (tile & 15) << 4;
    int y  = (tile >> 4) & 255;
    int b  = tile >> 12;
    int m  = lane & 15, hi = lane >> 4;
    int xx = x0 + m;                                     // this lane's A row (pixel)

    v8f acc0 = {}, acc1 = {}, acc2 = {}, acc3 = {};
    union AF { v16h v; uint4 u[2]; };
    const v16h* wbase = (const v16h*)w2p;

#pragma unroll
    for (int t = 0; t < 9; ++t) {
        int ky = t / 3, kx = t % 3;
        int sy = y + ky - 1;
        int sx = xx + kx - 1;
        AF a;
        if (sx >= 0 && sx < 256 && sy >= 0 && sy < 256) {
            // pixel base (64B aligned) + hi*8 halfs: two 16B loads give K hi*8.. and 16+hi*8..
            const uint4* p = (const uint4*)(x1n + ((((size_t)(b * 256 + sy)) * 256 + sx) << 5) + hi * 8);
            a.u[0] = p[0];
            a.u[1] = p[2];
        } else {
            a.u[0] = make_uint4(0, 0, 0, 0);
            a.u[1] = make_uint4(0, 0, 0, 0);
        }
        const v16h* wp = wbase + (size_t)(t * 4) * 32 + lane;   // + nt*32 per N-tile
        v16h bm0 = wp[0];
        acc0 = __builtin_amdgcn_wmma_f32_16x16x32_f16(false, a.v, false, bm0, (short)0, acc0, false, false);
        v16h bm1 = wp[32];
        acc1 = __builtin_amdgcn_wmma_f32_16x16x32_f16(false, a.v, false, bm1, (short)0, acc1, false, false);
        v16h bm2 = wp[64];
        acc2 = __builtin_amdgcn_wmma_f32_16x16x32_f16(false, a.v, false, bm2, (short)0, acc2, false, false);
        v16h bm3 = wp[96];
        acc3 = __builtin_amdgcn_wmma_f32_16x16x32_f16(false, a.v, false, bm3, (short)0, acc3, false, false);
    }

    // C/D layout: lane -> N = lane&15; reg r -> M = r + 8*hi. Store NHWC f32 + bias.
    int n = lane & 15;
    size_t pixbase = ((size_t)(b * 256 + y) * 256 + x0) * 64;
#pragma unroll
    for (int r = 0; r < 8; ++r) {
        size_t base = pixbase + (size_t)(r + 8 * hi) * 64;
        x2[base +      n] = acc0[r] + sb2[n];
        x2[base + 16 + n] = acc1[r] + sb2[16 + n];
        x2[base + 32 + n] = acc2[r] + sb2[32 + n];
        x2[base + 48 + n] = acc3[r] + sb2[48 + n];
    }
}

// ---------------------------------------------------------------- BN2 + ReLU + 2x2 maxpool -> NCHW out
__global__ void k_bn2pool(const float* __restrict__ x2, const float* __restrict__ g,
                          const float* __restrict__ beta, const float* __restrict__ st,
                          float* __restrict__ out) {
    size_t idx = (size_t)blockIdx.x * 256 + threadIdx.x; // 8*64*128*128 exact
    int ox = (int)(idx & 127);
    int oy = (int)((idx >> 7) & 127);
    int c  = (int)((idx >> 14) & 63);
    int b  = (int)(idx >> 20);
    float mean = st[c] * (1.0f / NSTAT);
    float var  = st[64 + c] * (1.0f / NSTAT) - mean * mean;
    float sc = g[c] * rsqrtf(var + 1e-5f);
    float sh = beta[c] - mean * sc;
    size_t base = ((size_t)(b * 256 + oy * 2) * 256 + ox * 2) * 64 + c;
    float v00 = fmaxf(fmaf(x2[base],             sc, sh), 0.0f);
    float v01 = fmaxf(fmaf(x2[base + 64],        sc, sh), 0.0f);
    float v10 = fmaxf(fmaf(x2[base + 64 * 256],  sc, sh), 0.0f);
    float v11 = fmaxf(fmaf(x2[base + 64 * 256 + 64], sc, sh), 0.0f);
    out[idx] = fmaxf(fmaxf(v00, v01), fmaxf(v10, v11));
}

// ================================================================ host launcher
extern "C" void kernel_launch(void* const* d_in, const int* in_sizes, int n_in,
                              void* d_out, int out_size, void* d_ws, size_t ws_size,
                              hipStream_t stream) {
    const float* img   = (const float*)d_in[0];
    const float* w1    = (const float*)d_in[1];
    const float* b1    = (const float*)d_in[2];
    const float* g1    = (const float*)d_in[3];
    const float* beta1 = (const float*)d_in[4];
    const float* w2    = (const float*)d_in[5];
    const float* b2    = (const float*)d_in[6];
    const float* g2    = (const float*)d_in[7];
    const float* beta2 = (const float*)d_in[8];
    float* out = (float*)d_out;

    char* ws = (char*)d_ws;
    size_t off = 0;
    auto take = [&](size_t bytes) { char* p = ws + off; off += (bytes + 255) & ~(size_t)255; return p; };
    float*    q    = (float*)take((size_t)NPIX * 4);
    float*    filt = (float*)take((size_t)NPIX * 4);
    float*    lbp  = (float*)take((size_t)NPIX * 4);
    float*    x1   = (float*)take((size_t)NPIX * 32 * 4);
    _Float16* x1n  = (_Float16*)take((size_t)NPIX * 32 * 2);
    _Float16* w2p  = (_Float16*)take((size_t)9 * 4 * 32 * 16 * 2);
    float*    st1  = (float*)take(64 * 4);
    float*    x2   = (float*)take((size_t)NPIX * 64 * 4);
    float*    st2  = (float*)take(128 * 4);

    k_gray<<<NPIX / 256, 256, 0, stream>>>(img, q);
    k_bilateral<<<dim3(16, 16, 8), dim3(16, 16), 0, stream>>>(q, filt);
    k_lbp<<<NPIX / 256, 256, 0, stream>>>(filt, lbp);

    k_zero_stats<<<1, 256, 0, stream>>>(st1, st2);
    k_conv1<<<NPIX / 256, 256, 0, stream>>>(lbp, w1, b1, x1);
    k_stats<<<2048, 256, 0, stream>>>(x1, st1, st1 + 32, 32, (size_t)NPIX * 32);
    k_bnrelu1<<<(unsigned)((size_t)NPIX * 32 / 256), 256, 0, stream>>>(x1, g1, beta1, st1, x1n);

    k_packw2<<<5, 256, 0, stream>>>(w2, w2p);
    k_conv2_wmma<<<4096, 256, 0, stream>>>(x1n, w2p, b2, x2);
    k_stats<<<2048, 256, 0, stream>>>(x2, st2, st2 + 64, 64, (size_t)NPIX * 64);

    k_bn2pool<<<(unsigned)((size_t)B_ * 64 * 128 * 128 / 256), 256, 0, stream>>>(x2, g2, beta2, st2, out);
}
